// EncoderLayer_32624571580900
// MI455X (gfx1250) — compile-verified, matrix operand order checked by reading, not run on hardware
//
#include <hip/hip_runtime.h>
#include <hip/hip_bf16.h>

// ---------------------------------------------------------------------------
// MI455X (gfx1250) implementation.
// All GEMMs run on v_wmma_f32_16x16x32_bf16 (bf16 in, f32 accum).
// Block tile 256x128x32 staged through LDS, 8 wave32 waves per block (4x2),
// 64x64 per wave = 16 WMMA tiles -> 16 wmma per 16 ds_load_b128 per K-step
// (vs 8:12 at 32x64), better matched to the WMMA pipe vs LDS read bandwidth.
// FFT band-split is folded into a circulant 512x512 GEMM; y_high = x1 - y_low.
// ---------------------------------------------------------------------------

typedef __attribute__((ext_vector_type(16))) __bf16 v16bf;
typedef __attribute__((ext_vector_type(8)))  __bf16 bf16x8;
typedef __attribute__((ext_vector_type(8)))  float  v8f;

union FragU { v16bf v; bf16x8 h[2]; };

#define BM 256
#define BN 128
#define BKK 32
#define SK 40   // padded LDS row stride (bf16 elems): 80B, keeps 16B alignment

enum {
  EPI_F32        = 0,  // Cf = acc + bias
  EPI_RELU_BF16  = 1,  // Cb = bf16(relu(acc + bias))
  EPI_BF16       = 2,  // Cb = bf16(acc + bias)
  EPI_RELU2_F32  = 3,  // Cf = relu(acc)^2   (attention scores)
  EPI_SIG_BF16   = 4,  // Cb = bf16(sigmoid(acc + bias))
  EPI_TRANS_BF16 = 6,  // Cb[(b*N+gn)<<lbSh | l] = bf16(acc+bias)  (V transposed)
  EPI_FFT        = 7   // Cb = bf16(acc); Cb2 = bf16(Rf - acc)     (band split)
};

template<int EPI>
__global__ __launch_bounds__(256)
void gemm_bf16_wmma(const __bf16* __restrict__ A,   // [M x K] row-major
                    const __bf16* __restrict__ Bt,  // [N x K] row-major (B^T)
                    const float*  __restrict__ bias,
                    float*  __restrict__ Cf,
                    __bf16* __restrict__ Cb,
                    __bf16* __restrict__ Cb2,
                    const float* __restrict__ Rf,
                    int M, int N, int K,
                    long long strideA, long long strideB, long long strideC,
                    int lbSh)
{
  __shared__ __align__(16) __bf16 sA[BM * SK];   // 256x32 tile, 20.0KB
  __shared__ __align__(16) __bf16 sB[BN * SK];   // 128x32 tile, 10.0KB

  const int tid  = threadIdx.x;
  const int wave = tid >> 5;
  const int lane = tid & 31;
  const int r    = lane & 15;   // row/col within 16x16 tile
  const int hi   = lane >> 4;   // selects K-halves per ISA 16-bit layout
  const int wm   = (wave & 3) * 64;   // 4 waves along M (256 rows)
  const int wn   = (wave >> 2) * 64;  // 2 waves along N (128 cols)
  const int bm   = blockIdx.y * BM;
  const int bn   = blockIdx.x * BN;
  const int z    = blockIdx.z;

  const __bf16* Az = A  + (long long)z * strideA;
  const __bf16* Bz = Bt + (long long)z * strideB;

  v8f acc[4][4] = {};

  const int lrow = tid >> 1;        // 0..127
  const int lcol = (tid & 1) * 16;  // 0 or 16

  for (int k0 = 0; k0 < K; k0 += BKK) {
    // ---- stage 256x32 A tile (two row-passes) + 128x32 B tile into LDS
#pragma unroll
    for (int rp = 0; rp < 2; rp++) {
      const int row = rp * 128 + lrow;
      const __bf16* ga = Az + (long long)(bm + row) * K + (k0 + lcol);
      *(bf16x8*)&sA[row * SK + lcol]     = *(const bf16x8*)(ga);
      *(bf16x8*)&sA[row * SK + lcol + 8] = *(const bf16x8*)(ga + 8);
    }
    {
      const __bf16* gb = Bz + (long long)(bn + lrow) * K + (k0 + lcol);
      *(bf16x8*)&sB[lrow * SK + lcol]     = *(const bf16x8*)(gb);
      *(bf16x8*)&sB[lrow * SK + lcol + 8] = *(const bf16x8*)(gb + 8);
    }
    __syncthreads();

    FragU aF[4], bF[4];
#pragma unroll
    for (int i = 0; i < 4; i++) {
      const int row = wm + i * 16 + r;
      aF[i].h[0] = *(const bf16x8*)&sA[row * SK + hi * 8];
      aF[i].h[1] = *(const bf16x8*)&sA[row * SK + 16 + hi * 8];
    }
#pragma unroll
    for (int j = 0; j < 4; j++) {
      const int col = wn + j * 16 + r;
      bF[j].h[0] = *(const bf16x8*)&sB[col * SK + hi * 8];
      bF[j].h[1] = *(const bf16x8*)&sB[col * SK + 16 + hi * 8];
    }

#pragma unroll
    for (int i = 0; i < 4; i++)
#pragma unroll
      for (int j = 0; j < 4; j++)
        acc[i][j] = __builtin_amdgcn_wmma_f32_16x16x32_bf16(
            false, aF[i].v, false, bF[j].v, (short)0, acc[i][j], false, false);
    __syncthreads();
  }

  // Epilogue. C layout: lane group n = r, VGPR e -> m = hi*8 + e.
  const long long co = strideC * (long long)z;
#pragma unroll
  for (int i = 0; i < 4; i++) {
#pragma unroll
    for (int j = 0; j < 4; j++) {
      const int gn = bn + wn + j * 16 + r;
      const float bv = bias ? bias[gn] : 0.0f;
#pragma unroll
      for (int e = 0; e < 8; e++) {
        const int gm = bm + wm + i * 16 + hi * 8 + e;
        const float v = acc[i][j][e] + bv;
        if constexpr (EPI == EPI_F32) {
          Cf[co + (long long)gm * N + gn] = v;
        } else if constexpr (EPI == EPI_RELU_BF16) {
          Cb[co + (long long)gm * N + gn] = (__bf16)fmaxf(v, 0.0f);
        } else if constexpr (EPI == EPI_BF16) {
          Cb[co + (long long)gm * N + gn] = (__bf16)v;
        } else if constexpr (EPI == EPI_RELU2_F32) {
          const float t = fmaxf(v, 0.0f);
          Cf[co + (long long)gm * N + gn] = t * t;
        } else if constexpr (EPI == EPI_SIG_BF16) {
          Cb[co + (long long)gm * N + gn] = (__bf16)(1.0f / (1.0f + __expf(-v)));
        } else if constexpr (EPI == EPI_TRANS_BF16) {
          const int b = gm >> lbSh, l = gm & ((1 << lbSh) - 1);
          Cb[(((long long)b * N + gn) << lbSh) + l] = (__bf16)v;
        } else if constexpr (EPI == EPI_FFT) {
          const long long idx = co + (long long)gm * N + gn;
          const float x1v = Rf[idx];
          Cb[idx]  = (__bf16)v;
          Cb2[idx] = (__bf16)(x1v - v);
        }
      }
    }
  }
}

// Row normalization of ReLU^2 scores: w = s / (sum + 1e-9); emits f32 (output)
// and bf16 (for the attn @ V WMMA GEMM).
__global__ __launch_bounds__(256)
void attn_norm_kernel(float* __restrict__ S, __bf16* __restrict__ Sb, int Lrow)
{
  __shared__ float red[256];
  const long long row = blockIdx.x;
  float* p = S + row * Lrow;
  float s = 0.0f;
  for (int i = threadIdx.x; i < Lrow; i += 256) s += p[i];
  red[threadIdx.x] = s;
  __syncthreads();
  for (int o = 128; o > 0; o >>= 1) {
    if ((int)threadIdx.x < o) red[threadIdx.x] += red[threadIdx.x + o];
    __syncthreads();
  }
  const float inv = 1.0f / (red[0] + 1e-9f);
  for (int i = threadIdx.x; i < Lrow; i += 256) {
    const float w = p[i] * inv;
    p[i] = w;
    Sb[row * Lrow + i] = (__bf16)w;
  }
}

// LayerNorm over D=512 with 0..2 residual adds; 256 threads, 2 elems/thread.
template<int NRES>
__global__ __launch_bounds__(256)
void layernorm_kernel(const float* __restrict__ X,
                      const float* __restrict__ R1,
                      const float* __restrict__ R2,
                      const float* __restrict__ g,
                      const float* __restrict__ bta,
                      float* __restrict__ Of, __bf16* __restrict__ Ob, int Dd)
{
  __shared__ float red[256];
  __shared__ float red2[256];
  const long long row = blockIdx.x;
  float v[2];
  float s1 = 0.0f, s2 = 0.0f;
#pragma unroll
  for (int t = 0; t < 2; t++) {
    const int i = threadIdx.x + t * 256;
    float val = X[row * Dd + i];
    if (NRES >= 1) val += R1[row * Dd + i];
    if (NRES >= 2) val += R2[row * Dd + i];
    v[t] = val; s1 += val; s2 += val * val;
  }
  red[threadIdx.x] = s1; red2[threadIdx.x] = s2;
  __syncthreads();
  for (int o = 128; o > 0; o >>= 1) {
    if ((int)threadIdx.x < o) {
      red[threadIdx.x]  += red[threadIdx.x + o];
      red2[threadIdx.x] += red2[threadIdx.x + o];
    }
    __syncthreads();
  }
  const float mean = red[0] / (float)Dd;
  const float var  = red2[0] / (float)Dd - mean * mean;
  const float rstd = rsqrtf(var + 1e-5f);
#pragma unroll
  for (int t = 0; t < 2; t++) {
    const int i = threadIdx.x + t * 256;
    const float y = (v[t] - mean) * rstd * g[i] + bta[i];
    Of[row * Dd + i] = y;
    if (Ob) Ob[row * Dd + i] = (__bf16)y;
  }
}

__global__ void fuse_kernel(const __bf16* __restrict__ gl, const __bf16* __restrict__ gh,
                            const __bf16* __restrict__ yl, const __bf16* __restrict__ yh,
                            __bf16* __restrict__ ylf, __bf16* __restrict__ yhf, long long n)
{
  const long long i = (long long)blockIdx.x * 256 + threadIdx.x;
  if (i < n) {
    const float G1 = (float)gl[i], G2 = (float)gh[i];
    const float a = (float)yl[i], b = (float)yh[i];
    ylf[i] = (__bf16)(G1 * a + (1.0f - G1) * b);
    yhf[i] = (__bf16)(G2 * b + (1.0f - G2) * a);
  }
}

__global__ void f2bf_kernel(const float* __restrict__ in, __bf16* __restrict__ out, long long n)
{
  const long long i = (long long)blockIdx.x * 256 + threadIdx.x;
  if (i < n) out[i] = (__bf16)in[i];
}

// h[d] = (1/D) * sum_{k in low band} cos(2*pi*k*d/D); band = [0,e) U [D-e,D).
__global__ void circ_h_kernel(float* __restrict__ h)
{
  const int Dd = 512, e = 64;
  const int d = threadIdx.x;
  const float w = 6.2831853071795864769f / (float)Dd;
  float s = 0.0f;
  for (int k = 0; k < e; k++)        s += __cosf(w * (float)((k * d) & (Dd - 1)));
  for (int k = Dd - e; k < Dd; k++)  s += __cosf(w * (float)((k * d) & (Dd - 1)));
  h[d] = s / (float)Dd;
}

// Bt[o][d] = h[(o - d) mod D]  (B^T rows for y_low = x1 @ C)
__global__ void circ_build_kernel(const float* __restrict__ h, __bf16* __restrict__ C)
{
  const int o = blockIdx.x, d = threadIdx.x;
  C[o * 512 + d] = (__bf16)h[(o - d + 512) & 511];
}

// ---------------------------------------------------------------------------

#define LAUNCH_GEMM(EPI, A_, B_, bias_, Cf_, Cb_, Cb2_, Rf_, M_, N_, K_, sA_, sB_, sC_, Z_, Lb_) \
  gemm_bf16_wmma<EPI><<<dim3((unsigned)((N_) / BN), (unsigned)((M_) / BM), (unsigned)(Z_)), 256, 0, stream>>>( \
      (const __bf16*)(A_), (const __bf16*)(B_), (const float*)(bias_),        \
      (float*)(Cf_), (__bf16*)(Cb_), (__bf16*)(Cb2_), (const float*)(Rf_),    \
      (M_), (N_), (K_), (long long)(sA_), (long long)(sB_), (long long)(sC_), (Lb_))

extern "C" void kernel_launch(void* const* d_in, const int* in_sizes, int n_in,
                              void* d_out, int out_size, void* d_ws, size_t ws_size,
                              hipStream_t stream)
{
  (void)in_sizes; (void)n_in; (void)out_size; (void)ws_size;
  const int B = 8, L = 2048, D = 512, DFF = 2048;
  const long long BLD  = (long long)B * L * D;     // 8.39M
  const long long BLL  = (long long)B * L * L;     // 33.5M
  const long long BLF  = (long long)B * L * DFF;   // 33.5M
  const long long DDl  = (long long)D * D;
  const long long DFFD = (long long)DFF * D;
  const int M = B * L;                             // 16384

  // ---- inputs (setup_inputs dict order: x, then params in insertion order)
  const float* x          = (const float*)d_in[0];
  const float* q_w = (const float*)d_in[1],  * q_b = (const float*)d_in[2];
  const float* k_w = (const float*)d_in[3],  * k_b = (const float*)d_in[4];
  const float* v_w = (const float*)d_in[5],  * v_b = (const float*)d_in[6];
  const float* c1_low_w  = (const float*)d_in[7],  * c1_low_b  = (const float*)d_in[8];
  const float* c2_low_w  = (const float*)d_in[9],  * c2_low_b  = (const float*)d_in[10];
  const float* c1_high_w = (const float*)d_in[11], * c1_high_b = (const float*)d_in[12];
  const float* c2_high_w = (const float*)d_in[13], * c2_high_b = (const float*)d_in[14];
  const float* c1_lo_w   = (const float*)d_in[15], * c1_lo_b   = (const float*)d_in[16];
  const float* c2_lo_w   = (const float*)d_in[17], * c2_lo_b   = (const float*)d_in[18];
  const float* c1_hig_w  = (const float*)d_in[19], * c1_hig_b  = (const float*)d_in[20];
  const float* c2_hig_w  = (const float*)d_in[21], * c2_hig_b  = (const float*)d_in[22];
  const float* gl_w = (const float*)d_in[23], * gl_b = (const float*)d_in[24];
  const float* gh_w = (const float*)d_in[25], * gh_b = (const float*)d_in[26];
  const float* n1_g = (const float*)d_in[27], * n1_b = (const float*)d_in[28];
  const float* n2_g = (const float*)d_in[29], * n2_b = (const float*)d_in[30];

  float* out_x    = (float*)d_out;         // [B,L,D]
  float* out_attn = out_x + BLD;           // [B,L,L]

  // ---- workspace carve-out
  char* p = (char*)d_ws;
  auto alloc = [&](long long bytes) -> void* {
    void* r = (void*)p;
    p += (bytes + 255) & ~255LL;
    return r;
  };
  __bf16* xb     = (__bf16*)alloc(BLD * 2);
  __bf16* Qb     = (__bf16*)alloc(BLD * 2);
  __bf16* Kb     = (__bf16*)alloc(BLD * 2);
  __bf16* Vtb    = (__bf16*)alloc(BLD * 2);   // [B][D][L]
  __bf16* attnb  = (__bf16*)alloc(BLL * 2);
  float*  newx   = (float*) alloc(BLD * 4);
  float*  x1     = (float*) alloc(BLD * 4);
  __bf16* x1b    = (__bf16*)alloc(BLD * 2);
  __bf16* ylb    = (__bf16*)alloc(BLD * 2);
  __bf16* yhb    = (__bf16*)alloc(BLD * 2);
  __bf16* hb     = (__bf16*)alloc(BLF * 2);   // FFN hidden, reused 4x
  __bf16* yl1b   = (__bf16*)alloc(BLD * 2);
  __bf16* yh1b   = (__bf16*)alloc(BLD * 2);
  __bf16* glb    = (__bf16*)alloc(BLD * 2);
  __bf16* ghb    = (__bf16*)alloc(BLD * 2);
  __bf16* ylfb   = (__bf16*)alloc(BLD * 2);
  __bf16* yhfb   = (__bf16*)alloc(BLD * 2);
  float*  ylow2  = (float*) alloc(BLD * 4);
  float*  yhigh2 = (float*) alloc(BLD * 4);
  __bf16* qwb    = (__bf16*)alloc(DDl * 2);
  __bf16* kwb    = (__bf16*)alloc(DDl * 2);
  __bf16* vwb    = (__bf16*)alloc(DDl * 2);
  __bf16* c1lowb = (__bf16*)alloc(DFFD * 2);
  __bf16* c2lowb = (__bf16*)alloc(DFFD * 2);
  __bf16* c1highb= (__bf16*)alloc(DFFD * 2);
  __bf16* c2highb= (__bf16*)alloc(DFFD * 2);
  __bf16* c1lob  = (__bf16*)alloc(DFFD * 2);
  __bf16* c2lob  = (__bf16*)alloc(DFFD * 2);
  __bf16* c1higb = (__bf16*)alloc(DFFD * 2);
  __bf16* c2higb = (__bf16*)alloc(DFFD * 2);
  __bf16* glwb   = (__bf16*)alloc(DDl * 2);
  __bf16* ghwb   = (__bf16*)alloc(DDl * 2);
  __bf16* circb  = (__bf16*)alloc(DDl * 2);
  float*  hcoef  = (float*) alloc(512 * 4);

  auto cvt = [&](const float* src, __bf16* dst, long long n) {
    f2bf_kernel<<<dim3((unsigned)((n + 255) / 256)), 256, 0, stream>>>(src, dst, n);
  };

  // ---- conversions + circulant operator
  cvt(x, xb, BLD);
  cvt(q_w, qwb, DDl);  cvt(k_w, kwb, DDl);  cvt(v_w, vwb, DDl);
  cvt(c1_low_w,  c1lowb,  DFFD);  cvt(c2_low_w,  c2lowb,  DFFD);
  cvt(c1_high_w, c1highb, DFFD);  cvt(c2_high_w, c2highb, DFFD);
  cvt(c1_lo_w,   c1lob,   DFFD);  cvt(c2_lo_w,   c2lob,   DFFD);
  cvt(c1_hig_w,  c1higb,  DFFD);  cvt(c2_hig_w,  c2higb,  DFFD);
  cvt(gl_w, glwb, DDl);  cvt(gh_w, ghwb, DDl);
  circ_h_kernel<<<1, 512, 0, stream>>>(hcoef);
  circ_build_kernel<<<512, 512, 0, stream>>>(hcoef, circb);

  // ---- attention
  LAUNCH_GEMM(EPI_BF16,       xb, qwb, q_b, nullptr, Qb,  nullptr, nullptr, M, D, D, 0, 0, 0, 1, 0);
  LAUNCH_GEMM(EPI_BF16,       xb, kwb, k_b, nullptr, Kb,  nullptr, nullptr, M, D, D, 0, 0, 0, 1, 0);
  LAUNCH_GEMM(EPI_TRANS_BF16, xb, vwb, v_b, nullptr, Vtb, nullptr, nullptr, M, D, D, 0, 0, 0, 1, 11);
  // scores = relu(Q K^T)^2 per batch, written straight into the attn output
  LAUNCH_GEMM(EPI_RELU2_F32, Qb, Kb, nullptr, out_attn, nullptr, nullptr, nullptr,
              L, L, D, (long long)L * D, (long long)L * D, (long long)L * L, B, 0);
  attn_norm_kernel<<<(unsigned)(B * L), 256, 0, stream>>>(out_attn, attnb, L);
  // new_x = attn @ V  (B^T rows = V transposed)
  LAUNCH_GEMM(EPI_F32, attnb, Vtb, nullptr, newx, nullptr, nullptr, nullptr,
              L, D, L, (long long)L * L, (long long)D * L, (long long)L * D, B, 0);
  layernorm_kernel<1><<<(unsigned)(B * L), 256, 0, stream>>>(x, newx, nullptr, n1_g, n1_b, x1, x1b, D);

  // ---- FFT band split as circulant GEMM: ylb = bf16(x1 @ C), yhb = bf16(x1 - x1@C)
  LAUNCH_GEMM(EPI_FFT, x1b, circb, nullptr, nullptr, ylb, yhb, x1, M, D, D, 0, 0, 0, 1, 0);

  // ---- first dual FFN
  LAUNCH_GEMM(EPI_RELU_BF16, ylb, c1lowb,  c1_low_b,  nullptr, hb,   nullptr, nullptr, M, DFF, D, 0, 0, 0, 1, 0);
  LAUNCH_GEMM(EPI_BF16,      hb,  c2lowb,  c2_low_b,  nullptr, yl1b, nullptr, nullptr, M, D, DFF, 0, 0, 0, 1, 0);
  LAUNCH_GEMM(EPI_RELU_BF16, yhb, c1highb, c1_high_b, nullptr, hb,   nullptr, nullptr, M, DFF, D, 0, 0, 0, 1, 0);
  LAUNCH_GEMM(EPI_BF16,      hb,  c2highb, c2_high_b, nullptr, yh1b, nullptr, nullptr, M, D, DFF, 0, 0, 0, 1, 0);

  // ---- gated cross-guidance fusion
  LAUNCH_GEMM(EPI_SIG_BF16, yl1b, glwb, gl_b, nullptr, glb, nullptr, nullptr, M, D, D, 0, 0, 0, 1, 0);
  LAUNCH_GEMM(EPI_SIG_BF16, yh1b, ghwb, gh_b, nullptr, ghb, nullptr, nullptr, M, D, D, 0, 0, 0, 1, 0);
  fuse_kernel<<<(unsigned)(BLD / 256), 256, 0, stream>>>(glb, ghb, yl1b, yh1b, ylfb, yhfb, BLD);

  // ---- second dual FFN
  LAUNCH_GEMM(EPI_RELU_BF16, ylfb, c1lob,  c1_lo_b,  nullptr, hb,     nullptr, nullptr, M, DFF, D, 0, 0, 0, 1, 0);
  LAUNCH_GEMM(EPI_F32,       hb,   c2lob,  c2_lo_b,  ylow2,   nullptr, nullptr, nullptr, M, D, DFF, 0, 0, 0, 1, 0);
  LAUNCH_GEMM(EPI_RELU_BF16, yhfb, c1higb, c1_hig_b, nullptr, hb,     nullptr, nullptr, M, DFF, D, 0, 0, 0, 1, 0);
  LAUNCH_GEMM(EPI_F32,       hb,   c2higb, c2_hig_b, yhigh2,  nullptr, nullptr, nullptr, M, D, DFF, 0, 0, 0, 1, 0);

  // ---- final residual LayerNorm -> output
  layernorm_kernel<2><<<(unsigned)(B * L), 256, 0, stream>>>(x1, ylow2, yhigh2, n2_g, n2_b, out_x, nullptr, D);
}
